// Attention_317827580753
// MI455X (gfx1250) — matrix operand order repaired and checked by reading, not compiled, not run on hardware
//
#include <hip/hip_runtime.h>
#include <hip/hip_bf16.h>

// ---------- types ----------
typedef __bf16 bf16_t;
typedef __attribute__((ext_vector_type(16))) __bf16 v16bf;
typedef __attribute__((ext_vector_type(8)))  __bf16 v8bf;
typedef __attribute__((ext_vector_type(8)))  float   v8f;
typedef __attribute__((ext_vector_type(4)))  unsigned int u32x4;
typedef __attribute__((ext_vector_type(8)))  int     i32x8;
typedef __attribute__((ext_vector_type(4)))  int     i32x4;

#define L_SEQ   4096
#define D_HID   1024
#define N_HEADS 16
#define HD      64
// ln(10000)/32
#define ROPE_C  0.2878231366242588f

// ---------------------------------------------------------------
// Stage 0: fp32 -> bf16 conversion
// ---------------------------------------------------------------
__global__ void cvt_f32_bf16(const float* __restrict__ s, bf16_t* __restrict__ d, int n) {
    int i = blockIdx.x * blockDim.x + threadIdx.x;
    if (i < n) d[i] = (bf16_t)s[i];
}

// ---------------------------------------------------------------
// Shared GEMM strip:  C(16x64) = A(16x1024) @ W^T  via WMMA bf16
//   Fragment layouts used here (A/B in contiguous k order, same for both):
//     A 16x32 (16-bit): lane holds row M=lane%16; elem e -> K=e+16*hi
//     B 32x16 (16-bit): lane holds col N=lane%16; elem e -> K=e+16*hi
//     C 16x16 f32     : elem r -> M=r+8*hi, N=lane%16
// ---------------------------------------------------------------
__device__ __forceinline__ void gemm_strip(const bf16_t* __restrict__ A,
                                           const bf16_t* __restrict__ W,
                                           int l0, int n0, int lane, v8f c[4]) {
    const int li = lane & 15;
    const int hi = (lane >> 4) & 1;
    c[0] = v8f{}; c[1] = v8f{}; c[2] = v8f{}; c[3] = v8f{};
    const bf16_t* arow = A + (l0 + li) * D_HID;
    for (int k0 = 0; k0 < D_HID; k0 += 32) {
        union { v16bf v; v8bf h[2]; } a;
        a.h[0] = *(const v8bf*)(arow + k0 + 16 * hi);
        a.h[1] = *(const v8bf*)(arow + k0 + 16 * hi + 8);
#pragma unroll
        for (int j = 0; j < 4; ++j) {
            const bf16_t* wrow = W + (n0 + j * 16 + li) * D_HID + k0 + 16 * hi;
            v16bf b = *(const v16bf*)wrow;
            c[j] = __builtin_amdgcn_wmma_f32_16x16x32_bf16(
                       false, a.v, false, b, (short)0, c[j], false, false);
        }
    }
}

// ---------------------------------------------------------------
// TDM helper: DMA a 2-D bf16 tile (tile_d0 x tile_d1, row stride
// stride_d0 elements) from global into LDS at lds_byte_off.
// D# bit layout per CDNA5 ISA sections 8.3 - 8.6.
// ---------------------------------------------------------------
__device__ __forceinline__ void tdm_load_tile_2d(unsigned int lds_byte_off,
                                                 const void* gaddr,
                                                 unsigned int tile_d0,
                                                 unsigned int tile_d1,
                                                 unsigned long long stride_d0) {
    unsigned long long ga = (unsigned long long)(uintptr_t)gaddr;
    u32x4 g0;
    g0[0] = 1u;                                            // count=1, user mode
    g0[1] = lds_byte_off;                                  // lds_addr (bytes)
    g0[2] = (unsigned int)ga;                              // global_addr[31:0]
    g0[3] = (unsigned int)((ga >> 32) & 0x01ffffffu)       // global_addr[56:32]
          | (2u << 30);                                    // type=2 (image)
    i32x8 g1;
    g1[0] = (int)(1u << 16);                               // wg_mask=0, data_size=1 (2B)
    g1[1] = (int)((tile_d0 & 0xffffu) << 16);              // tensor_dim0 = tile_d0
    g1[2] = (int)((tile_d0 >> 16) | ((tile_d1 & 0xffffu) << 16)); // tensor_dim1 = tile_d1
    g1[3] = (int)((tile_d1 >> 16) | ((tile_d0 & 0xffffu) << 16)); // tile_dim0
    g1[4] = (int)(tile_d1 & 0xffffu);                      // tile_dim1, tile_dim2=0
    g1[5] = (int)(unsigned int)(stride_d0 & 0xffffffffu);  // tensor_dim0_stride lo
    g1[6] = (int)(unsigned int)((stride_d0 >> 32) & 0xffffu); // stride hi, dim1_stride=0
    g1[7] = 0;
    i32x4 g2;
    g2[0] = 1;                                             // tensor_dim2 = 1
    g2[1] = 1;                                             // tensor_dim3 = 1
    g2[2] = 0;                                             // tensor_dim2_stride
    g2[3] = 0;                                             // tile_dim3 = 0
    i32x4 g3;
    g3[0] = 0;                                             // tensor_dim3_stride
    g3[1] = (int)(1u << 16);                               // tensor_dim4 = 1
    g3[2] = 0;                                             // tile_dim4 = 0
    g3[3] = 0;
#if defined(__clang_major__) && (__clang_major__ >= 23)
    __builtin_amdgcn_tensor_load_to_lds(g0, g1, g2, g3, i32x8{}, 0);
#else
    __builtin_amdgcn_tensor_load_to_lds(g0, g1, g2, g3, 0);
#endif
}

// ---------------------------------------------------------------
// Stage 1: QKV projection + RoPE (Q,K) / transpose-store (V)
//   grid.x = 512 (8 waves/block, 4096 strips), grid.z = 0:Q 1:K 2:V
//   Q,K out: [H][L][64] bf16 (RoPE applied);  V out: [H][64][L] bf16
// ---------------------------------------------------------------
__global__ __launch_bounds__(256)
void qkv_rope_kernel(const bf16_t* __restrict__ xb,
                     const bf16_t* __restrict__ wqb,
                     const bf16_t* __restrict__ wkb,
                     const bf16_t* __restrict__ wvb,
                     bf16_t* __restrict__ qb,
                     bf16_t* __restrict__ kb,
                     bf16_t* __restrict__ vt) {
    const int lane = threadIdx.x & 31;
    const int wave = threadIdx.x >> 5;
    const int strip = blockIdx.x * 8 + wave;          // 4096 strips
    const int l0 = (strip >> 4) * 16;                 // 256 row tiles
    const int n0 = (strip & 15) * 64;                 // 16 col strips (one head each)
    const int mode = blockIdx.z;
    const bf16_t* W = (mode == 0) ? wqb : (mode == 1) ? wkb : wvb;

    v8f c[4];
    gemm_strip(xb, W, l0, n0, lane, c);

    const int li = lane & 15;
    const int hi = (lane >> 4) & 1;
    const int h  = n0 >> 6;

    if (mode < 2) {
        bf16_t* dst = (mode == 0) ? qb : kb;
        float fj[4];
#pragma unroll
        for (int j = 0; j < 4; ++j) {
            int d = j * 16 + li;                       // 0..63 within head
            fj[j] = __expf(-ROPE_C * (float)(d & 31)); // inv_freq[d%32]
        }
#pragma unroll
        for (int r = 0; r < 8; ++r) {
            const int   lrow = l0 + r + 8 * hi;
            const float pos  = (float)lrow;
            float v0 = c[0][r], v1 = c[1][r], v2 = c[2][r], v3 = c[3][r];
            float o0, o1, o2, o3;
            {   float a = pos * fj[0]; o0 = cosf(a) * v0 - sinf(a) * v2; }
            {   float a = pos * fj[1]; o1 = cosf(a) * v1 - sinf(a) * v3; }
            {   float a = pos * fj[2]; o2 = cosf(a) * v2 + sinf(a) * v0; }
            {   float a = pos * fj[3]; o3 = cosf(a) * v3 + sinf(a) * v1; }
            bf16_t* row = dst + ((size_t)h * L_SEQ + lrow) * HD + li;
            row[0]  = (bf16_t)o0;
            row[16] = (bf16_t)o1;
            row[32] = (bf16_t)o2;
            row[48] = (bf16_t)o3;
        }
    } else {
        // V: store transposed vt[h][d][l]
#pragma unroll
        for (int j = 0; j < 4; ++j) {
            const int d = j * 16 + li;
#pragma unroll
            for (int r = 0; r < 8; ++r) {
                const int lrow = l0 + r + 8 * hi;
                vt[((size_t)h * HD + d) * L_SEQ + lrow] = (bf16_t)c[j][r];
            }
        }
    }
}

// ---------------------------------------------------------------
// Stage 2: causal flash attention with TDM-staged K/V in LDS
//   grid.x = H * (L/64) = 1024 blocks, 128 threads (4 waves).
//   Wave w owns the 16-row Q tile q0 = block_base + 16*w.
//   Per 32-key chunk: wave 0 DMAs K (32x64) and V (64x32, from the
//   pre-transposed vt) tiles into LDS (double-buffered, TENSORcnt),
//   all 4 waves consume them via ds loads + WMMA.
// ---------------------------------------------------------------
__global__ __launch_bounds__(128)
void flash_attn_kernel(const bf16_t* __restrict__ qb,
                       const bf16_t* __restrict__ kb,
                       const bf16_t* __restrict__ vt,
                       bf16_t* __restrict__ ao) {
    __shared__ bf16_t Ks[2][32 * 64];                  // [buf][kpos][d]   4KB each
    __shared__ bf16_t Vs[2][64 * 32];                  // [buf][d][kpos]   4KB each
    __shared__ bf16_t Pst[4][16 * 32];                 // per-wave P staging

    const int lane = threadIdx.x & 31;
    const int wave = threadIdx.x >> 5;
    const int li = lane & 15;
    const int hi = (lane >> 4) & 1;
    const int h      = blockIdx.x >> 6;
    const int q_base = (blockIdx.x & 63) * 64;
    const int q0     = q_base + wave * 16;
    const int kend   = q_base + 64;                    // uniform causal extent

    const bf16_t* Q  = qb + (size_t)h * L_SEQ * HD;
    const bf16_t* Kg = kb + (size_t)h * L_SEQ * HD;
    const bf16_t* Vg = vt + (size_t)h * HD * L_SEQ;

    // Q A-fragments, d 0..31 and 32..63
    union { v16bf v; v8bf p[2]; } qa[2];
    const bf16_t* qrow = Q + (q0 + li) * HD;
    qa[0].p[0] = *(const v8bf*)(qrow + 16 * hi);
    qa[0].p[1] = *(const v8bf*)(qrow + 8 + 16 * hi);
    qa[1].p[0] = *(const v8bf*)(qrow + 32 + 16 * hi);
    qa[1].p[1] = *(const v8bf*)(qrow + 40 + 16 * hi);

    v8f o[4];
    o[0] = v8f{}; o[1] = v8f{}; o[2] = v8f{}; o[3] = v8f{};
    float m[8], l[8];
#pragma unroll
    for (int r = 0; r < 8; ++r) { m[r] = -3.0e38f; l[r] = 0.0f; }

    // prologue: DMA chunk 0 into buffer 0 (wave 0 only)
    if (wave == 0) {
        tdm_load_tile_2d((unsigned int)(uintptr_t)(const void*)&Ks[0][0],
                         Kg, HD, 32, HD);
        tdm_load_tile_2d((unsigned int)(uintptr_t)(const void*)&Vs[0][0],
                         Vg, 32, HD, L_SEQ);
    }

    bf16_t* lp = Pst[wave];
    int buf = 0;
    for (int k0 = 0; k0 < kend; k0 += 32) {
        const bool has_next = (k0 + 32) < kend;
        __syncthreads();                               // prev reads of buf^1 done
        if (wave == 0) {
            if (has_next) {
                tdm_load_tile_2d((unsigned int)(uintptr_t)(const void*)&Ks[buf ^ 1][0],
                                 Kg + (size_t)(k0 + 32) * HD, HD, 32, HD);
                tdm_load_tile_2d((unsigned int)(uintptr_t)(const void*)&Vs[buf ^ 1][0],
                                 Vg + (k0 + 32), 32, HD, L_SEQ);
                __builtin_amdgcn_s_wait_tensorcnt(2);  // chunk k0 complete
            } else {
                __builtin_amdgcn_s_wait_tensorcnt(0);
            }
        }
        __syncthreads();                               // buf chunk visible to all

        if (k0 < q0 + 16) {                            // causal work for this wave
            const bf16_t* KsB = &Ks[buf][0];
            const bf16_t* VsB = &Vs[buf][0];
            // S tiles: s[g] covers key columns k0+g*16 .. +15
            v8f s[2]; s[0] = v8f{}; s[1] = v8f{};
#pragma unroll
            for (int g = 0; g < 2; ++g) {
                const bf16_t* krow = KsB + (g * 16 + li) * HD + 16 * hi;
#pragma unroll
                for (int dg = 0; dg < 2; ++dg) {
                    v16bf b = *(const v16bf*)(krow + dg * 32);
                    s[g] = __builtin_amdgcn_wmma_f32_16x16x32_bf16(
                               false, qa[dg].v, false, b, (short)0, s[g], false, false);
                }
            }
            // scale + causal mask
#pragma unroll
            for (int g = 0; g < 2; ++g)
#pragma unroll
                for (int r = 0; r < 8; ++r) {
                    const int kg = k0 + g * 16 + li;
                    const int qg = q0 + r + 8 * hi;
                    float v = s[g][r] * 0.125f;        // 1/sqrt(64)
                    s[g][r] = (kg > qg) ? -3.0e38f : v;
                }
            // online softmax: row reductions across the 16 lanes of each row
            float t[8];
#pragma unroll
            for (int r = 0; r < 8; ++r) t[r] = fmaxf(s[0][r], s[1][r]);
#pragma unroll
            for (int msk = 1; msk <= 8; msk <<= 1)
#pragma unroll
                for (int r = 0; r < 8; ++r) t[r] = fmaxf(t[r], __shfl_xor(t[r], msk, 32));
            float al[8];
#pragma unroll
            for (int r = 0; r < 8; ++r) {
                float mn = fmaxf(m[r], t[r]);
                al[r] = __expf(m[r] - mn);
                m[r] = mn;
            }
#pragma unroll
            for (int g = 0; g < 2; ++g)
#pragma unroll
                for (int r = 0; r < 8; ++r) s[g][r] = __expf(s[g][r] - m[r]);
            float rs[8];
#pragma unroll
            for (int r = 0; r < 8; ++r) rs[r] = s[0][r] + s[1][r];
#pragma unroll
            for (int msk = 1; msk <= 8; msk <<= 1)
#pragma unroll
                for (int r = 0; r < 8; ++r) rs[r] += __shfl_xor(rs[r], msk, 32);
#pragma unroll
            for (int r = 0; r < 8; ++r) l[r] = l[r] * al[r] + rs[r];
#pragma unroll
            for (int j = 0; j < 4; ++j)
#pragma unroll
                for (int r = 0; r < 8; ++r) o[j][r] *= al[r];

            // transpose P (C-layout) -> A-layout via per-wave LDS staging
#pragma unroll
            for (int g = 0; g < 2; ++g)
#pragma unroll
                for (int r = 0; r < 8; ++r)
                    lp[(r + 8 * hi) * 32 + g * 16 + li] = (bf16_t)s[g][r];
            asm volatile("s_wait_dscnt 0" ::: "memory"); // same-wave LDS store->load
            union { v16bf v; v8bf p[2]; } pa;
            pa.p[0] = *(const v8bf*)(lp + li * 32 + 16 * hi);
            pa.p[1] = *(const v8bf*)(lp + li * 32 + 8 + 16 * hi);

            // O += P @ V  (Vs: [d][kpos], per-lane contiguous in kpos)
#pragma unroll
            for (int j = 0; j < 4; ++j) {
                v16bf vb = *(const v16bf*)(VsB + (j * 16 + li) * 32 + 16 * hi);
                o[j] = __builtin_amdgcn_wmma_f32_16x16x32_bf16(
                           false, pa.v, false, vb, (short)0, o[j], false, false);
            }
        }
        buf ^= 1;
    }

    // epilogue: normalize and store head-interleaved output [L][1024]
#pragma unroll
    for (int r = 0; r < 8; ++r) {
        const float inv = 1.0f / l[r];
        const int qrow_i = q0 + r + 8 * hi;
        bf16_t* dst = ao + (size_t)qrow_i * D_HID + h * HD + li;
        dst[0]  = (bf16_t)(o[0][r] * inv);
        dst[16] = (bf16_t)(o[1][r] * inv);
        dst[32] = (bf16_t)(o[2][r] * inv);
        dst[48] = (bf16_t)(o[3][r] * inv);
    }
}

// ---------------------------------------------------------------
// Stage 3: output projection  out = ao @ wo^T  (fp32 store)
// ---------------------------------------------------------------
__global__ __launch_bounds__(256)
void out_proj_kernel(const bf16_t* __restrict__ ao,
                     const bf16_t* __restrict__ wob,
                     float* __restrict__ out) {
    const int lane = threadIdx.x & 31;
    const int wave = threadIdx.x >> 5;
    const int strip = blockIdx.x * 8 + wave;
    const int l0 = (strip >> 4) * 16;
    const int n0 = (strip & 15) * 64;
    v8f c[4];
    gemm_strip(ao, wob, l0, n0, lane, c);
    const int li = lane & 15;
    const int hi = (lane >> 4) & 1;
#pragma unroll
    for (int j = 0; j < 4; ++j)
#pragma unroll
        for (int r = 0; r < 8; ++r)
            out[(size_t)(l0 + r + 8 * hi) * D_HID + n0 + j * 16 + li] = c[j][r];
}

// ---------------------------------------------------------------
extern "C" void kernel_launch(void* const* d_in, const int* in_sizes, int n_in,
                              void* d_out, int out_size, void* d_ws, size_t ws_size,
                              hipStream_t stream) {
    const float* x  = (const float*)d_in[0];
    const float* wq = (const float*)d_in[1];
    const float* wk = (const float*)d_in[2];
    const float* wv = (const float*)d_in[3];
    const float* wo = (const float*)d_in[4];
    float* out = (float*)d_out;

    char* ws = (char*)d_ws;
    const size_t NX = (size_t)L_SEQ * D_HID;    // 4M elements
    const size_t NW = (size_t)D_HID * D_HID;    // 1M elements
    size_t off = 0;
    bf16_t* xb  = (bf16_t*)(ws + off); off += NX * 2;
    bf16_t* wqb = (bf16_t*)(ws + off); off += NW * 2;
    bf16_t* wkb = (bf16_t*)(ws + off); off += NW * 2;
    bf16_t* wvb = (bf16_t*)(ws + off); off += NW * 2;
    bf16_t* wob = (bf16_t*)(ws + off); off += NW * 2;
    bf16_t* qb  = (bf16_t*)(ws + off); off += NX * 2;
    bf16_t* kb  = (bf16_t*)(ws + off); off += NX * 2;
    bf16_t* vt  = (bf16_t*)(ws + off); off += NX * 2;
    bf16_t* ao  = (bf16_t*)(ws + off); off += NX * 2;

    // Stage 0: bf16 conversions
    cvt_f32_bf16<<<(int)((NX + 255) / 256), 256, 0, stream>>>(x,  xb,  (int)NX);
    cvt_f32_bf16<<<(int)((NW + 255) / 256), 256, 0, stream>>>(wq, wqb, (int)NW);
    cvt_f32_bf16<<<(int)((NW + 255) / 256), 256, 0, stream>>>(wk, wkb, (int)NW);
    cvt_f32_bf16<<<(int)((NW + 255) / 256), 256, 0, stream>>>(wv, wvb, (int)NW);
    cvt_f32_bf16<<<(int)((NW + 255) / 256), 256, 0, stream>>>(wo, wob, (int)NW);

    // Stage 1: QKV projections + RoPE (z = 0:Q, 1:K, 2:V)
    qkv_rope_kernel<<<dim3(512, 1, 3), 256, 0, stream>>>(xb, wqb, wkb, wvb, qb, kb, vt);

    // Stage 2: causal flash attention (TDM-staged K/V)
    flash_attn_kernel<<<N_HEADS * (L_SEQ / 64), 128, 0, stream>>>(qb, kb, vt, ao);

    // Stage 3: output projection (fp32)
    out_proj_kernel<<<512, 256, 0, stream>>>(ao, wob, out);
}
